// MultiheadAttention_89223650607335
// MI455X (gfx1250) — matrix operand order repaired and checked by reading, not compiled, not run on hardware
//
#include <hip/hip_runtime.h>
#include <hip/hip_bf16.h>

// ---------------------------------------------------------------------------
// MI455X (gfx1250) multi-head attention, bf16 WMMA + f32 accumulate,
// TDM (tensor_load_to_lds) staging of GEMM B-panels.
//   B=2, S=2048, D=1024, H=16, HD=64
// Workspace layout (bytes):
//   [0,        8M)  xb   : x in bf16 [4096][1024]  (reused as attn-out later)
//   [8M,      10M)  wqb  : Wq bf16 [1024][1024] row-major
//   [10M,     12M)  wkb
//   [12M,     14M)  wvb
//   [14M,     16M)  wob
//   [16M,     24M)  qb   : [B,H,S,64] bf16
//   [24M,     32M)  kb
//   [32M,     40M)  vb
// ---------------------------------------------------------------------------

typedef __attribute__((ext_vector_type(16))) __bf16 v16bf;
typedef __attribute__((ext_vector_type(8)))  __bf16 v8bf;
typedef __attribute__((ext_vector_type(8)))  float  v8f;
typedef __attribute__((ext_vector_type(4)))  unsigned int v4u;
typedef __attribute__((ext_vector_type(8)))  int v8i;
typedef __attribute__((ext_vector_type(4)))  int v4i;

#define WMMA_BF16(A, B, C) \
  __builtin_amdgcn_wmma_f32_16x16x32_bf16(false, (A), false, (B), (short)0, (C), false, false)

#if __has_builtin(__builtin_amdgcn_tensor_load_to_lds)
#define ATHENA_HAS_TDM 1
#else
#define ATHENA_HAS_TDM 0
#endif

constexpr int BATCH = 2;
constexpr int SEQ   = 2048;
constexpr int DM    = 1024;
constexpr int NH    = 16;
constexpr int HDIM  = 64;
constexpr int NTOK  = BATCH * SEQ;   // 4096

// Load a 16-bit A-fragment half-row: elems 0..7 from p[0..7], elems 8..15 from p[16..23]
// (A layout: lane half h -> K = 8h + j (j<8), K = 16 + 8h + (j-8) (j>=8))
__device__ __forceinline__ v16bf ld_afrag(const __bf16* p) {
  v8bf lo = *(const v8bf*)p;
  v8bf hi = *(const v8bf*)(p + 16);
  v16bf r;
#pragma unroll
  for (int i = 0; i < 8; ++i) { r[i] = lo[i]; r[i + 8] = hi[i]; }
  return r;
}

#if ATHENA_HAS_TDM
// Issue a TDM load of a 2D bf16 tile: 32 rows x 64 cols, row stride 1024 elems,
// from global `gsrc` into LDS byte offset `lds_off`. D# per ISA 08_async_tensor.
//   group0: [1:0]=count=1, [63:32]=lds_addr, [120:64]=global_addr, [127:126]=type=2
//   group1: [17:16]=data_size=1 (2B), [79:48]=tensor_dim0, [111:80]=tensor_dim1,
//           [127:112]=tile_dim0, [143:128]=tile_dim1, [207:160]=tensor_dim0_stride
// 6-arg builtin form (clang-23 / therock-10.0):
//   (uint32x4 g0, int32x8 g1, int32x4 g2, int32x4 g3, int32x8 pad, i32 cpol)
__device__ __forceinline__ void tdm_load_b_panel(const __bf16* gsrc, unsigned lds_off) {
  unsigned long long ga = (unsigned long long)(uintptr_t)gsrc;
  v4u g0 = { 1u,                                   // count=1, user descriptor
             lds_off,                              // lds_addr
             (unsigned)ga,                         // global_addr[31:0]
             (unsigned)(ga >> 32) | (2u << 30) };  // global_addr[56:32] | type=2
  v8i g1 = { (int)(1u << 16),          // data_size = 1 (2 bytes)
             (int)(1024u << 16),       // tensor_dim0[15:0]=1024 (bits 63:48)
             (int)(2048u << 16),       // tensor_dim0[31:16]=0 | tensor_dim1[15:0]=2048
             (int)(64u << 16),         // tensor_dim1[31:16]=0 | tile_dim0=64
             (int)32,                  // tile_dim1=32, tile_dim2=0
             (int)1024,                // tensor_dim0_stride[31:0]=1024
             0, 0 };                   // stride hi / dim1_stride = 0
  v4i z4 = { 0, 0, 0, 0 };
  v8i z8 = { 0, 0, 0, 0, 0, 0, 0, 0 };
  __builtin_amdgcn_tensor_load_to_lds(g0, g1, z4, z4, z8, 0);
}
#endif

// ---------------------------------------------------------------------------
// fp32 -> bf16 conversion (RNE via hardware convert)
// ---------------------------------------------------------------------------
__global__ __launch_bounds__(256) void cvt_f32_bf16(const float* __restrict__ s,
                                                    __bf16* __restrict__ d, int n) {
  int i = (blockIdx.x * 256 + threadIdx.x) * 4;
  if (i + 3 < n) {
    float4 v = *(const float4*)(s + i);
    d[i + 0] = (__bf16)v.x;
    d[i + 1] = (__bf16)v.y;
    d[i + 2] = (__bf16)v.z;
    d[i + 3] = (__bf16)v.w;
  }
}

// ---------------------------------------------------------------------------
// GEMM: C[M=4096][N=1024] = A[4096][1024] * Bw[1024][1024] + bias
// Block = 128 threads = 4 waves; each wave owns a 32x32 tile (2x2 WMMA tiles).
// B-panel (32k x 64n, 4KB) staged per block via TDM into LDS, double-buffered;
// TENSORcnt pipelining: issue slab ki+1, wait<=1 so slab ki is resident.
// QKV=true : output scattered to bf16 [B,H,S,64]
// QKV=false: output fp32 row-major [M][N]  (final projection into d_out)
// ---------------------------------------------------------------------------
template <bool QKV>
__global__ __launch_bounds__(128) void gemm_bf16(const __bf16* __restrict__ A,
                                                 const __bf16* __restrict__ Bw,
                                                 const float* __restrict__ bias,
                                                 void* __restrict__ Cout) {
  const int lane  = threadIdx.x & 31;
  const int wave  = threadIdx.x >> 5;
  const int half  = lane >> 4;
  const int l15   = lane & 15;
  const int wn    = wave & 1;
  const int Mtile = blockIdx.y * 64 + (wave >> 1) * 32;
  const int Nblk  = blockIdx.x * 64;
  constexpr int K = 1024;
  constexpr int NSLAB = K / 32;  // 32 k-slabs

  v8f acc[2][2] = {};

#if ATHENA_HAS_TDM
  __shared__ alignas(32) __bf16 bpanel[2][32 * 64];  // 2 x 4KB
  const __bf16* bblock = Bw + Nblk;                  // row k at bblock + k*DM
  const unsigned lds0 = (unsigned)(uintptr_t)&bpanel[0][0];
  const unsigned lds1 = (unsigned)(uintptr_t)&bpanel[1][0];

  if (wave == 0) tdm_load_b_panel(bblock, lds0);     // slab 0 -> buf0

  for (int ki = 0; ki < NSLAB; ++ki) {
    const int k0 = ki * 32;
    if (wave == 0) {
      if (ki + 1 < NSLAB) {
        tdm_load_b_panel(bblock + (size_t)(k0 + 32) * DM, (ki & 1) ? lds0 : lds1);
        __builtin_amdgcn_s_wait_tensorcnt(1);  // slab ki resident (in-order TDM)
      } else {
        __builtin_amdgcn_s_wait_tensorcnt(0);
      }
    }
    __syncthreads();  // slab ki visible to all waves

    v16bf af[2];
#pragma unroll
    for (int sm = 0; sm < 2; ++sm) {
      const __bf16* ap = A + (size_t)(Mtile + sm * 16 + l15) * K + k0 + half * 8;
      af[sm] = ld_afrag(ap);
      __builtin_prefetch(ap + 32, 0, 1);  // global_prefetch_b8, next K slab
    }
    const __bf16* bp = (ki & 1) ? &bpanel[1][0] : &bpanel[0][0];
    v16bf bfr[2];
#pragma unroll
    for (int sn = 0; sn < 2; ++sn) {
      // B layout: lane&15 = N column, elems = K rows (8h+s / 16+8h+s-8) -> strided LDS gather
      for (int s = 0; s < 16; ++s)
        bfr[sn][s] = bp[((s & 7) + 8 * half + ((s >> 3) << 4)) * 64 + wn * 32 + sn * 16 + l15];
    }
#pragma unroll
    for (int sm = 0; sm < 2; ++sm)
#pragma unroll
      for (int sn = 0; sn < 2; ++sn)
        acc[sm][sn] = WMMA_BF16(af[sm], bfr[sn], acc[sm][sn]);

    __syncthreads();  // all reads of slab ki done before its buffer is reloaded
  }
#else
  // Fallback: direct-global B fragments (L2-resident on 192MB L2)
  for (int k0 = 0; k0 < K; k0 += 32) {
    v16bf af[2];
#pragma unroll
    for (int sm = 0; sm < 2; ++sm) {
      const __bf16* ap = A + (size_t)(Mtile + sm * 16 + l15) * K + k0 + half * 8;
      af[sm] = ld_afrag(ap);
      __builtin_prefetch(ap + 32, 0, 1);
    }
    v16bf bfr[2];
#pragma unroll
    for (int sn = 0; sn < 2; ++sn)
      for (int s = 0; s < 16; ++s)
        bfr[sn][s] = Bw[(size_t)(k0 + (s & 7) + 8 * half + ((s >> 3) << 4)) * DM + Nblk + wn * 32 + sn * 16 + l15];
#pragma unroll
    for (int sm = 0; sm < 2; ++sm)
#pragma unroll
      for (int sn = 0; sn < 2; ++sn)
        acc[sm][sn] = WMMA_BF16(af[sm], bfr[sn], acc[sm][sn]);
  }
#endif

  // Epilogue. C layout: elem r -> M = r + 8*half, lane l15 -> N.
  const int Ntile = Nblk + wn * 32;
  if constexpr (QKV) {
    __bf16* out = (__bf16*)Cout;
#pragma unroll
    for (int sn = 0; sn < 2; ++sn) {
      const int n  = Ntile + sn * 16 + l15;
      const float bv = bias[n];
      const int h = n >> 6, hd = n & 63;
#pragma unroll
      for (int sm = 0; sm < 2; ++sm)
#pragma unroll
        for (int r = 0; r < 8; ++r) {
          const int m = Mtile + sm * 16 + r + 8 * half;
          const int b = m >> 11, sidx = m & (SEQ - 1);
          out[(((size_t)(b * NH + h) * SEQ + sidx) << 6) + hd] =
              (__bf16)(acc[sm][sn][r] + bv);
        }
    }
  } else {
    float* out = (float*)Cout;
#pragma unroll
    for (int sn = 0; sn < 2; ++sn) {
      const int n  = Ntile + sn * 16 + l15;
      const float bv = bias[n];
#pragma unroll
      for (int sm = 0; sm < 2; ++sm)
#pragma unroll
        for (int r = 0; r < 8; ++r) {
          const int m = Mtile + sm * 16 + r + 8 * half;
          out[(size_t)m * DM + n] = acc[sm][sn][r] + bv;
        }
    }
  }
}

// ---------------------------------------------------------------------------
// Flash attention with ALiBi + causal mask.
// One wave per (b, h, 16-query tile). Computes S^T = K * Q^T so that:
//   * score C-layout: elem r = key, lane&15 = query  -> softmax rows on lanes
//   * post-softmax P^T in C-layout IS the A-fragment layout of P (no movement)
//   * V loads are strided B-fragment gathers (lane&15 = hd col, elems = keys)
// ---------------------------------------------------------------------------
__global__ __launch_bounds__(128) void attn_alibi(const __bf16* __restrict__ qg,
                                                  const __bf16* __restrict__ kg,
                                                  const __bf16* __restrict__ vg,
                                                  __bf16* __restrict__ attnb) {
  const int lane = threadIdx.x & 31;
  const int wave = threadIdx.x >> 5;
  const int half = lane >> 4;
  const int l15  = lane & 15;

  const int task = blockIdx.x * 4 + wave;  // 4096 tasks = 2*16*128
  const int qt   = task & 127;             // query tile (16 rows)
  const int bh   = task >> 7;              // 0..31
  const int h    = bh & (NH - 1);
  const int b    = bh >> 4;

  const __bf16* qb = qg + (size_t)bh * SEQ * HDIM;
  const __bf16* kb = kg + (size_t)bh * SEQ * HDIM;
  const __bf16* vb = vg + (size_t)bh * SEQ * HDIM;

  const float slope = exp2f(-0.5f * (float)(h + 1));  // (2^{-8/16})^{h+1}
  const float inv_sqrt_hd = 0.125f;
  const int   qpos = qt * 16 + l15;                   // this lane's query position

  // Q^T B-fragments, loaded once: chunk c covers hd in [32c, 32c+32).
  // B layout: lane&15 = N (query), elems = k (hd) in table order.
  v16bf qf[2];
#pragma unroll
  for (int c = 0; c < 2; ++c) {
    qf[c] = ld_afrag(qb + (size_t)(qt * 16 + l15) * HDIM + 32 * c + 8 * half);
  }

  float mrow = -1e30f, lrow = 0.0f;
  v8f   oacc[4] = {};

  const int npairs = (qt >> 1) + 1;  // key pairs of 32 covering [0, qt*16+16)
  for (int p = 0; p < npairs; ++p) {
    const int kk = p * 32;

    // Scores S^T: A = K rows (16 keys x 64 hd, two 32-wide chunks)
    v8f sc[2] = {};
#pragma unroll
    for (int s = 0; s < 2; ++s) {
#pragma unroll
      for (int c = 0; c < 2; ++c) {
        const __bf16* kp =
            kb + (size_t)(kk + s * 16 + l15) * HDIM + 32 * c + 8 * half;
        v16bf kf = ld_afrag(kp);
        sc[s] = WMMA_BF16(kf, qf[c], sc[s]);
      }
    }

    // scale + alibi + causal; keys live on elements, query on lane
    float p0[8], p1[8];
    float tm = -1e30f;
#pragma unroll
    for (int r = 0; r < 8; ++r) {
      const int k0i = kk + r + 8 * half;
      const int k1i = k0i + 16;
      float s0 = sc[0][r] * inv_sqrt_hd + slope * (float)(k0i - qpos);
      float s1 = sc[1][r] * inv_sqrt_hd + slope * (float)(k1i - qpos);
      if (k0i > qpos) s0 = -1e30f;
      if (k1i > qpos) s1 = -1e30f;
      p0[r] = s0; p1[r] = s1;
      tm = fmaxf(tm, fmaxf(s0, s1));
    }
    tm = fmaxf(tm, __shfl_xor(tm, 16));
    const float mnew = fmaxf(mrow, tm);
    const float corr = __expf(mrow - mnew);
    float psum = 0.0f;
#pragma unroll
    for (int r = 0; r < 8; ++r) {
      p0[r] = __expf(p0[r] - mnew);
      p1[r] = __expf(p1[r] - mnew);
      psum += p0[r] + p1[r];
    }
    psum += __shfl_xor(psum, 16);
    lrow = lrow * corr + psum;
    mrow = mnew;

    // Rescale O: O-row M = r + 8*half; its state sits in lane (M | sameHalf)
    float f[8];
#pragma unroll
    for (int r = 0; r < 8; ++r)
      f[r] = __shfl(corr, (r + 8 * half) | (lane & 16), 32);
#pragma unroll
    for (int nt = 0; nt < 4; ++nt)
#pragma unroll
      for (int r = 0; r < 8; ++r) oacc[nt][r] *= f[r];

    // P^T C-layout == P A-fragment layout: direct repack
    v16bf pf;
#pragma unroll
    for (int r = 0; r < 8; ++r) {
      pf[r]     = (__bf16)p0[r];
      pf[r + 8] = (__bf16)p1[r];
    }

    // O += P * V; V B-fragment: lane&15 = hd column, elems = keys in table order
#pragma unroll
    for (int nt = 0; nt < 4; ++nt) {
      v16bf vf;
      for (int s = 0; s < 16; ++s)
        vf[s] = vb[(size_t)(kk + (s & 7) + 8 * half + ((s >> 3) << 4)) * HDIM + nt * 16 + l15];
      oacc[nt] = WMMA_BF16(pf, vf, oacc[nt]);
    }
  }

  // Normalize and scatter to attn-out [token][D] bf16 (col = h*64 + nt*16 + l15)
  const float linv = 1.0f / lrow;
  float g[8];
#pragma unroll
  for (int r = 0; r < 8; ++r)
    g[r] = __shfl(linv, (r + 8 * half) | (lane & 16), 32);
#pragma unroll
  for (int nt = 0; nt < 4; ++nt) {
    const int col = h * HDIM + nt * 16 + l15;
#pragma unroll
    for (int r = 0; r < 8; ++r) {
      const int srow = qt * 16 + r + 8 * half;
      attnb[(size_t)(b * SEQ + srow) * DM + col] = (__bf16)(oacc[nt][r] * g[r]);
    }
  }
}

// ---------------------------------------------------------------------------
extern "C" void kernel_launch(void* const* d_in, const int* in_sizes, int n_in,
                              void* d_out, int out_size, void* d_ws, size_t ws_size,
                              hipStream_t stream) {
  (void)in_sizes; (void)n_in; (void)out_size; (void)ws_size;
  const float* x  = (const float*)d_in[0];
  const float* Wq = (const float*)d_in[1];
  const float* bq = (const float*)d_in[2];
  const float* Wk = (const float*)d_in[3];
  const float* bk = (const float*)d_in[4];
  const float* Wv = (const float*)d_in[5];
  const float* bv = (const float*)d_in[6];
  const float* Wo = (const float*)d_in[7];
  const float* bo = (const float*)d_in[8];

  char* w = (char*)d_ws;
  const size_t MB = 1024 * 1024;
  __bf16* xb  = (__bf16*)(w + 0);        // 8 MB; reused as attention output
  __bf16* wqb = (__bf16*)(w + 8  * MB);  // 2 MB each
  __bf16* wkb = (__bf16*)(w + 10 * MB);
  __bf16* wvb = (__bf16*)(w + 12 * MB);
  __bf16* wob = (__bf16*)(w + 14 * MB);
  __bf16* qbf = (__bf16*)(w + 16 * MB);  // 8 MB each, [B,H,S,64]
  __bf16* kbf = (__bf16*)(w + 24 * MB);
  __bf16* vbf = (__bf16*)(w + 32 * MB);

  // 1) fp32 -> bf16 conversions
  cvt_f32_bf16<<<NTOK * DM / 1024, 256, 0, stream>>>(x, xb, NTOK * DM);
  cvt_f32_bf16<<<DM * DM / 1024, 256, 0, stream>>>(Wq, wqb, DM * DM);
  cvt_f32_bf16<<<DM * DM / 1024, 256, 0, stream>>>(Wk, wkb, DM * DM);
  cvt_f32_bf16<<<DM * DM / 1024, 256, 0, stream>>>(Wv, wvb, DM * DM);
  cvt_f32_bf16<<<DM * DM / 1024, 256, 0, stream>>>(Wo, wob, DM * DM);

  // 2) QKV projections (bf16 WMMA, TDM-staged B panels, head-major bf16 out)
  dim3 ggrid(DM / 64, NTOK / 64);  // (16, 64)
  gemm_bf16<true><<<ggrid, 128, 0, stream>>>(xb, wqb, bq, (void*)qbf);
  gemm_bf16<true><<<ggrid, 128, 0, stream>>>(xb, wkb, bk, (void*)kbf);
  gemm_bf16<true><<<ggrid, 128, 0, stream>>>(xb, wvb, bv, (void*)vbf);

  // 3) flash attention (alibi + causal); writes bf16 attn-out into xb
  attn_alibi<<<BATCH * NH * (SEQ / 16) / 4, 128, 0, stream>>>(qbf, kbf, vbf, xb);

  // 4) output projection -> fp32 d_out
  gemm_bf16<false><<<ggrid, 128, 0, stream>>>(xb, wob, bo, d_out);
}
